// MemLayer_46823733461178
// MI455X (gfx1250) — compile-verified, matrix operand order checked by reading, not run on hardware
//
#include <hip/hip_runtime.h>
#include <hip/hip_bf16.h>

typedef __attribute__((ext_vector_type(16))) _Float16 v16h;
typedef __attribute__((ext_vector_type(8)))  _Float16 v8h;
typedef __attribute__((ext_vector_type(8)))  float    v8f;
typedef __attribute__((ext_vector_type(4)))  unsigned v4u;
typedef __attribute__((ext_vector_type(4)))  int      v4i;
typedef __attribute__((ext_vector_type(8)))  int      v8i;

#define D_DIM 256
#define M_DIM 131072
#define B_DIM 512
#define C_DIM 100
#define K_TOP 256
#define MTILE 128
#define GCAP  2048

__device__ __forceinline__ unsigned f2key(float f) {
  unsigned u = __float_as_uint(f);
  return (u & 0x80000000u) ? ~u : (u | 0x80000000u);
}
__device__ __forceinline__ float key2f(unsigned k) {
  unsigned u = (k & 0x80000000u) ? (k & 0x7FFFFFFFu) : ~k;
  return __uint_as_float(u);
}

// One wave32 per row: L2-normalize 256 f32 -> 256 f16.
__global__ __launch_bounds__(256) void normalize_rows(const float* __restrict__ in,
                                                      _Float16* __restrict__ out,
                                                      int nrows) {
  int wv   = (int)(blockIdx.x * 256 + threadIdx.x) >> 5;
  int lane = threadIdx.x & 31;
  if (wv >= nrows) return;
  const float4* row = (const float4*)(in + (size_t)wv * D_DIM);
  float4 a = row[lane * 2];
  float4 b = row[lane * 2 + 1];
  float ss = a.x*a.x + a.y*a.y + a.z*a.z + a.w*a.w
           + b.x*b.x + b.y*b.y + b.z*b.z + b.w*b.w;
  #pragma unroll
  for (int off = 16; off > 0; off >>= 1) ss += __shfl_xor(ss, off, 32);
  float s = 1.0f / (sqrtf(ss) + 1e-12f);
  v8h o;
  o[0] = (_Float16)(a.x * s); o[1] = (_Float16)(a.y * s);
  o[2] = (_Float16)(a.z * s); o[3] = (_Float16)(a.w * s);
  o[4] = (_Float16)(b.x * s); o[5] = (_Float16)(b.y * s);
  o[6] = (_Float16)(b.z * s); o[7] = (_Float16)(b.w * s);
  *(v8h*)(out + (size_t)wv * D_DIM + lane * 8) = o;
}

// sims[B_DIM x M_DIM] = qn(BxD) * kn(MxD)^T via v_wmma_f32_16x16x32_f16.
// One workgroup per 128 key rows. The 64KB kn chunk is contiguous in memory,
// so it is staged into LDS with a single Tensor-Data-Mover DMA (1-D descriptor,
// 8192 x 8-byte elements), issued by wave 0 only (TDM ignores EXEC), tracked
// with TENSORcnt. B tiles then live in registers across the whole 512-query
// loop: keys touch HBM exactly once; qn (256KB) streams from L2.
__global__ __launch_bounds__(256) void gemm_f16_wmma(const _Float16* __restrict__ qn,
                                                     const _Float16* __restrict__ kn,
                                                     float* __restrict__ sims) {
  __shared__ _Float16 Bs[MTILE * D_DIM];  // 64 KB
  const int tid    = threadIdx.x;
  const int m_base = blockIdx.x * MTILE;

  if (tid < 32) {  // one wave issues the TDM transfer
    unsigned lds = (unsigned)(uintptr_t)(void*)Bs;  // low 32 bits = LDS offset
    unsigned long long ga =
        (unsigned long long)(uintptr_t)(kn + (size_t)m_base * D_DIM);
    v4u g0;
    g0.x = 1u;                                        // count=1, user mode
    g0.y = lds;                                       // lds_addr
    g0.z = (unsigned)(ga & 0xFFFFFFFFu);              // global_addr[31:0]
    g0.w = (unsigned)((ga >> 32) & 0x1FFFFFFu)        // global_addr[56:32]
         | (2u << 30);                                // type=2 (image)
    v8i g1;
    g1[0] = (int)(3u << 16);       // data_size = 8 bytes
    g1[1] = (int)(0x2000u << 16);  // tensor_dim0 = 8192 (low 16 bits)
    g1[2] = (int)(1u << 16);       // tensor_dim0 hi=0; tensor_dim1 = 1
    g1[3] = (int)(0x2000u << 16);  // tensor_dim1 hi=0; tile_dim0 = 8192
    g1[4] = 0;                     // tile_dim1/tile_dim2 unused
    g1[5] = 8192;                  // tensor_dim0_stride = 8192
    g1[6] = 0;
    g1[7] = 0;
    v4i gz4 = {0, 0, 0, 0};
    v8i gz8 = {0, 0, 0, 0, 0, 0, 0, 0};
    __builtin_amdgcn_tensor_load_to_lds(g0, g1, gz4, gz4, gz8, 0);
    __builtin_amdgcn_s_wait_tensorcnt(0);
  }
  __syncthreads();

  const int wave  = tid >> 5;
  const int lane  = tid & 31;
  const int col   = lane & 15;           // key column within the 16-wide tile
  const int khalf = (lane >> 4) * 8;     // lanes 16-31 take K+8 / K+24 chunks
  const int mloc  = wave * 16;           // 16 key columns per wave

  v16h Breg[8];
  {
    const _Float16* brow = Bs + (size_t)(mloc + col) * D_DIM + khalf;
    #pragma unroll
    for (int kk = 0; kk < 8; ++kk) {
      v8h lo = *(const v8h*)(brow + kk * 32);
      v8h hi = *(const v8h*)(brow + kk * 32 + 16);
      v16h t;
      #pragma unroll
      for (int j = 0; j < 8; ++j) { t[j] = lo[j]; t[j + 8] = hi[j]; }
      Breg[kk] = t;
    }
  }

  for (int qb = 0; qb < B_DIM; qb += 16) {
    const _Float16* arow = qn + (size_t)(qb + col) * D_DIM + khalf;
    v8f acc = {};
    #pragma unroll
    for (int kk = 0; kk < 8; ++kk) {
      v8h lo = *(const v8h*)(arow + kk * 32);
      v8h hi = *(const v8h*)(arow + kk * 32 + 16);
      v16h a;
      #pragma unroll
      for (int j = 0; j < 8; ++j) { a[j] = lo[j]; a[j + 8] = hi[j]; }
      acc = __builtin_amdgcn_wmma_f32_16x16x32_f16(false, a, false, Breg[kk],
                                                   (short)0, acc, false, false);
    }
    const int n     = m_base + mloc + col;
    const int rbase = qb + ((lane >> 4) * 8);   // D layout: rows r / r+8
    #pragma unroll
    for (int r = 0; r < 8; ++r)
      sims[(size_t)(rbase + r) * M_DIM + n] = acc[r];
  }
}

// One workgroup per query row: exact top-256 set via 3-level radix-select on a
// monotone float->uint key, index-ordered tie handling (matches lax.top_k),
// then argmax/softmax/class-sums/pos-neg maxes entirely in LDS.
__global__ __launch_bounds__(256) void topk_kernel(const float* __restrict__ sims,
                                                   const int* __restrict__ mem_vals,
                                                   const int* __restrict__ y,
                                                   float* __restrict__ out) {
  __shared__ unsigned hist[4096];
  __shared__ float    gval[GCAP];
  __shared__ int      gidx[GCAP];
  __shared__ unsigned gkey[GCAP];
  __shared__ float    classSum[C_DIM];
  __shared__ int      tiedIdx[64];
  __shared__ unsigned s_B, s_T;
  __shared__ int s_r, s_gcount, s_tcount, s_tieCut, s_y, s_argmax;
  __shared__ float s_vmax, s_Z;
  __shared__ unsigned s_posKey, s_negKey;
  __shared__ unsigned long long s_best;

  const int b   = blockIdx.x;
  const int tid = threadIdx.x;
  const float4* row4 = (const float4*)(sims + (size_t)b * M_DIM);

  for (int i = tid; i < 4096; i += 256) hist[i] = 0;
  for (int i = tid; i < C_DIM; i += 256) classSum[i] = 0.f;
  if (tid == 0) {
    s_gcount = 0; s_tcount = 0; s_y = y[b]; s_Z = 0.f;
    s_posKey = f2key(-2.0f); s_negKey = f2key(-2.0f); s_best = 0ull;
  }
  __syncthreads();

  // Level 1: 12-bit bucket histogram over the full row (vectorized b128 reads).
  for (int i = tid; i < M_DIM / 4; i += 256) {
    float4 v = row4[i];
    atomicAdd(&hist[f2key(v.x) >> 20], 1u);
    atomicAdd(&hist[f2key(v.y) >> 20], 1u);
    atomicAdd(&hist[f2key(v.z) >> 20], 1u);
    atomicAdd(&hist[f2key(v.w) >> 20], 1u);
  }
  __syncthreads();
  if (tid == 0) {
    unsigned c = 0;
    for (int bb = 4095; bb >= 0; --bb) {
      c += hist[bb];
      if (c >= (unsigned)K_TOP) {
        s_B = (unsigned)bb; s_r = K_TOP - (int)(c - hist[bb]); break;
      }
    }
  }
  __syncthreads();
  const unsigned Bsel = s_B;

  // Gather every candidate at or above the threshold bucket (expected ~360).
  for (int i = tid; i < M_DIM / 4; i += 256) {
    float4 v = row4[i];
    float vv[4] = {v.x, v.y, v.z, v.w};
    #pragma unroll
    for (int c4 = 0; c4 < 4; ++c4) {
      unsigned k = f2key(vv[c4]);
      if ((k >> 20) >= Bsel) {
        int p = atomicAdd(&s_gcount, 1);
        if (p < GCAP) { gval[p] = vv[c4]; gidx[p] = i * 4 + c4; gkey[p] = k; }
      }
    }
  }
  __syncthreads();
  const int gn = (s_gcount < GCAP) ? s_gcount : GCAP;

  // Level 2: bits [19:8] within the threshold bucket.
  for (int i = tid; i < 4096; i += 256) hist[i] = 0;
  __syncthreads();
  for (int j = tid; j < gn; j += 256)
    if ((gkey[j] >> 20) == Bsel) atomicAdd(&hist[(gkey[j] >> 8) & 0xFFFu], 1u);
  __syncthreads();
  if (tid == 0) {
    unsigned c = 0; int r = s_r; unsigned B2 = 0;
    for (int bb = 4095; bb >= 0; --bb) {
      c += hist[bb];
      if ((int)c >= r) { B2 = (unsigned)bb; s_r = r - (int)(c - hist[bb]); break; }
    }
    s_T = (Bsel << 20) | (B2 << 8);
  }
  __syncthreads();
  const unsigned T12 = s_T;

  // Level 3: final 8 bits -> exact 256th key.
  for (int i = tid; i < 256; i += 256) hist[i] = 0;
  __syncthreads();
  for (int j = tid; j < gn; j += 256)
    if ((gkey[j] >> 8) == (T12 >> 8)) atomicAdd(&hist[gkey[j] & 0xFFu], 1u);
  __syncthreads();
  if (tid == 0) {
    unsigned c = 0; int r = s_r; unsigned B3 = 0, e = 1;
    for (int bb = 255; bb >= 0; --bb) {
      c += hist[bb];
      if ((int)c >= r) { B3 = (unsigned)bb; e = hist[bb]; s_r = r - (int)(c - hist[bb]); break; }
    }
    s_T = T12 | B3;
    s_tieCut = ((unsigned)s_r == e) ? 0x7FFFFFFF : -2;  // -2 => resolve ties
  }
  __syncthreads();
  const unsigned T = s_T;

  if (s_tieCut == -2) {  // uniform branch: pick smallest-index equals (lax.top_k order)
    for (int j = tid; j < gn; j += 256)
      if (gkey[j] == T) { int p = atomicAdd(&s_tcount, 1); if (p < 64) tiedIdx[p] = gidx[j]; }
    __syncthreads();
    if (tid == 0) {
      int n = s_tcount;
      if (n > 64) s_tieCut = 0x7FFFFFFF;
      else {
        for (int i2 = 1; i2 < n; ++i2) {
          int v2 = tiedIdx[i2], j2 = i2 - 1;
          while (j2 >= 0 && tiedIdx[j2] > v2) { tiedIdx[j2 + 1] = tiedIdx[j2]; --j2; }
          tiedIdx[j2 + 1] = v2;
        }
        s_tieCut = tiedIdx[(s_r > 0 ? s_r : 1) - 1];
      }
    }
    __syncthreads();
  }
  const int tieCut = s_tieCut;

  // argmax over selected set (ties -> smallest index, matching top_k order).
  unsigned long long local = 0ull;
  for (int j = tid; j < gn; j += 256) {
    bool sel = (gkey[j] > T) || (gkey[j] == T && gidx[j] <= tieCut);
    if (sel) {
      unsigned long long pk = ((unsigned long long)gkey[j] << 32) |
                              (unsigned)(M_DIM - 1 - gidx[j]);
      if (pk > local) local = pk;
    }
  }
  atomicMax(&s_best, local);
  __syncthreads();
  if (tid == 0) {
    s_vmax   = key2f((unsigned)(s_best >> 32));
    s_argmax = M_DIM - 1 - (int)(s_best & 0xFFFFFFFFu);
  }
  __syncthreads();

  const float vmax = s_vmax;
  const int   yb   = s_y;
  for (int j = tid; j < gn; j += 256) {
    bool sel = (gkey[j] > T) || (gkey[j] == T && gidx[j] <= tieCut);
    if (sel) {
      float w = __expf(gval[j] - vmax);
      atomicAdd(&s_Z, w);
      int lbl = mem_vals[gidx[j]];
      atomicAdd(&classSum[lbl], w);
      if (lbl == yb) atomicMax(&s_posKey, gkey[j]);
      else           atomicMax(&s_negKey, gkey[j]);
    }
  }
  __syncthreads();

  float* out_closest = out;
  float* out_probs   = out + B_DIM;
  float* out_loss    = out + B_DIM + (size_t)B_DIM * C_DIM;
  if (tid < C_DIM) out_probs[(size_t)b * C_DIM + tid] = classSum[tid] / s_Z;
  if (tid == 0) {
    out_closest[b] = (float)mem_vals[s_argmax];
    float sp = key2f(s_posKey), sn = key2f(s_negKey);
    out_loss[b] = fmaxf(sn - sp + 0.1f, 0.0f);
  }
}

extern "C" void kernel_launch(void* const* d_in, const int* in_sizes, int n_in,
                              void* d_out, int out_size, void* d_ws, size_t ws_size,
                              hipStream_t stream) {
  (void)in_sizes; (void)n_in; (void)out_size; (void)ws_size;
  const float* x        = (const float*)d_in[0];
  const int*   y        = (const int*)d_in[1];
  const float* mem_keys = (const float*)d_in[2];
  const int*   mem_vals = (const int*)d_in[3];
  float* out = (float*)d_out;

  char* ws = (char*)d_ws;
  _Float16* qn = (_Float16*)(ws);                                       // 256 KB
  _Float16* kn = (_Float16*)(ws + (1u << 20));                          // 64 MB
  float* sims  = (float*)(ws + (1u << 20) + (size_t)M_DIM * D_DIM * 2); // 256 MB

  normalize_rows<<<B_DIM / 8, 256, 0, stream>>>(x, qn, B_DIM);
  normalize_rows<<<M_DIM / 8, 256, 0, stream>>>(mem_keys, kn, M_DIM);
  gemm_f16_wmma<<<M_DIM / MTILE, 256, 0, stream>>>(qn, kn, sims);
  topk_kernel<<<B_DIM, 256, 0, stream>>>(sims, mem_vals, y, out);
}
